// EncoderDecoderLSTM_69973607186724
// MI455X (gfx1250) — compile-verified, matrix operand order checked by reading, not run on hardware
//
#include <hip/hip_runtime.h>

// ---------------------------------------------------------------------------
// Encoder-decoder LSTM for MI455X (gfx1250, wave32, WMMA bf16).
// B=256, T=168, IN=64, H=512, L=2, HOR=24.
// Strategy: per (layer, timestep) fused gate kernel. Each wave computes one
// 16x16 (m, j) tile for ALL FOUR gates via v_wmma_f32_16x16x32_bf16, then
// applies sigmoid/tanh + cell update entirely in registers.
// ---------------------------------------------------------------------------

namespace {
constexpr int kB   = 256;
constexpr int kT   = 168;
constexpr int kIn  = 64;
constexpr int kH   = 512;
constexpr int kHor = 24;
constexpr int kG4  = 4 * kH;  // 2048
}

typedef __attribute__((ext_vector_type(16))) __bf16 v16bf;
typedef __attribute__((ext_vector_type(8)))  float  v8f;

__device__ __forceinline__ unsigned short f2bf(float f) {
  union { float f; unsigned u; } c; c.f = f;
  unsigned u = c.u;
  u = (u + 0x7FFFu + ((u >> 16) & 1u)) >> 16;   // round-to-nearest-even
  return (unsigned short)u;
}
__device__ __forceinline__ float bf2f(unsigned short b) {
  union { unsigned u; float f; } c; c.u = ((unsigned)b) << 16; return c.f;
}
__device__ __forceinline__ float sigmoidf_(float x) {
  return 1.0f / (1.0f + __expf(-x));
}

// ----------------------------- utility kernels -----------------------------

__global__ void cvt_f32_to_bf16(const float* __restrict__ src,
                                unsigned short* __restrict__ dst, int n) {
  int i = blockIdx.x * blockDim.x + threadIdx.x;
  if (i < n) dst[i] = f2bf(src[i]);
}

__global__ void zero_u32(unsigned* __restrict__ p, int n) {
  int i = blockIdx.x * blockDim.x + threadIdx.x;
  if (i < n) p[i] = 0u;
}

// ------------------------------- LSTM step ---------------------------------
// Computes, for one layer at one timestep:
//   G = Xa @ Wih^T + Hprev @ Whh^T + bias (+ xscalar * wih_scalar^T)
//   c' = sigmoid(G_f)*c + sigmoid(G_i)*tanh(G_g);  h' = sigmoid(G_o)*tanh(c')
// Grid: 128 blocks x 128 threads (4 waves). 512 waves total = 16 m-tiles x
// 32 j-tiles. Each wave owns a 16x16 (m,j) tile and all 4 gate columns.

__launch_bounds__(128)
__global__ void lstm_step(
    const unsigned short* __restrict__ Xa, int Kx, int ldx,   // bf16 [kB, Kx]
    const unsigned short* __restrict__ Wih,                   // bf16 [kG4, Kx]
    const unsigned short* __restrict__ Hprev,                 // bf16 [kB, kH]
    const unsigned short* __restrict__ Whh,                   // bf16 [kG4, kH]
    const float* __restrict__ bias,                           // f32 [kG4]
    const float* __restrict__ xscalar,                        // f32 [kB] or null
    const float* __restrict__ wih_scalar,                     // f32 [kG4] or null
    float* __restrict__ Cst,                                  // f32 [kB, kH] in/out
    unsigned short* __restrict__ Hout) {                      // bf16 [kB, kH]
  const int lane = threadIdx.x & 31;
  const int wave = threadIdx.x >> 5;
  const int gw   = blockIdx.x * 4 + wave;   // 0..511
  const int mt   = gw & 15;                 // 16 row tiles (M = 256)
  const int jt   = gw >> 4;                 // 32 col tiles (H = 512)
  const int r15  = lane & 15;
  const int koff = (lane >> 4) << 3;        // 0 or 8 (A/B half-wave K offset)

  v8f acc0 = {}, acc1 = {}, acc2 = {}, acc3 = {};

  union Tile { v16bf v; uint4 q[2]; };

  // ---- recurrent contribution: Hprev @ Whh^T ----
  {
    const unsigned short* arow = Hprev + (size_t)(mt * 16 + r15) * kH;
    const unsigned short* b0r  = Whh + (size_t)(0 * kH + jt * 16 + r15) * kH;
    const unsigned short* b1r  = Whh + (size_t)(1 * kH + jt * 16 + r15) * kH;
    const unsigned short* b2r  = Whh + (size_t)(2 * kH + jt * 16 + r15) * kH;
    const unsigned short* b3r  = Whh + (size_t)(3 * kH + jt * 16 + r15) * kH;
#pragma unroll 4
    for (int k = 0; k < kH; k += 32) {
      Tile A, Bt;
      A.q[0] = *(const uint4*)(arow + k + koff);
      A.q[1] = *(const uint4*)(arow + k + koff + 16);

      Bt.q[0] = *(const uint4*)(b0r + k + koff);
      Bt.q[1] = *(const uint4*)(b0r + k + koff + 16);
      acc0 = __builtin_amdgcn_wmma_f32_16x16x32_bf16(false, A.v, false, Bt.v,
                                                     (short)0, acc0, false, false);
      Bt.q[0] = *(const uint4*)(b1r + k + koff);
      Bt.q[1] = *(const uint4*)(b1r + k + koff + 16);
      acc1 = __builtin_amdgcn_wmma_f32_16x16x32_bf16(false, A.v, false, Bt.v,
                                                     (short)0, acc1, false, false);
      Bt.q[0] = *(const uint4*)(b2r + k + koff);
      Bt.q[1] = *(const uint4*)(b2r + k + koff + 16);
      acc2 = __builtin_amdgcn_wmma_f32_16x16x32_bf16(false, A.v, false, Bt.v,
                                                     (short)0, acc2, false, false);
      Bt.q[0] = *(const uint4*)(b3r + k + koff);
      Bt.q[1] = *(const uint4*)(b3r + k + koff + 16);
      acc3 = __builtin_amdgcn_wmma_f32_16x16x32_bf16(false, A.v, false, Bt.v,
                                                     (short)0, acc3, false, false);
    }
  }

  // ---- input contribution: Xa @ Wih^T (Kx = 64 or 512; skipped when 0) ----
  if (Kx > 0) {
    const unsigned short* arow = Xa + (size_t)(mt * 16 + r15) * ldx;
    const unsigned short* b0r  = Wih + (size_t)(0 * kH + jt * 16 + r15) * Kx;
    const unsigned short* b1r  = Wih + (size_t)(1 * kH + jt * 16 + r15) * Kx;
    const unsigned short* b2r  = Wih + (size_t)(2 * kH + jt * 16 + r15) * Kx;
    const unsigned short* b3r  = Wih + (size_t)(3 * kH + jt * 16 + r15) * Kx;
#pragma unroll 2
    for (int k = 0; k < Kx; k += 32) {
      Tile A, Bt;
      A.q[0] = *(const uint4*)(arow + k + koff);
      A.q[1] = *(const uint4*)(arow + k + koff + 16);

      Bt.q[0] = *(const uint4*)(b0r + k + koff);
      Bt.q[1] = *(const uint4*)(b0r + k + koff + 16);
      acc0 = __builtin_amdgcn_wmma_f32_16x16x32_bf16(false, A.v, false, Bt.v,
                                                     (short)0, acc0, false, false);
      Bt.q[0] = *(const uint4*)(b1r + k + koff);
      Bt.q[1] = *(const uint4*)(b1r + k + koff + 16);
      acc1 = __builtin_amdgcn_wmma_f32_16x16x32_bf16(false, A.v, false, Bt.v,
                                                     (short)0, acc1, false, false);
      Bt.q[0] = *(const uint4*)(b2r + k + koff);
      Bt.q[1] = *(const uint4*)(b2r + k + koff + 16);
      acc2 = __builtin_amdgcn_wmma_f32_16x16x32_bf16(false, A.v, false, Bt.v,
                                                     (short)0, acc2, false, false);
      Bt.q[0] = *(const uint4*)(b3r + k + koff);
      Bt.q[1] = *(const uint4*)(b3r + k + koff + 16);
      acc3 = __builtin_amdgcn_wmma_f32_16x16x32_bf16(false, A.v, false, Bt.v,
                                                     (short)0, acc3, false, false);
    }
  }

  // ---- epilogue: bias + (optional rank-1 input) + gates + cell update ----
  // C/D layout: lane in [0,16): VGPR e -> M = e; lane in [16,32): M = e + 8.
  // N = lane & 15 for all VGPRs.
  const int j = jt * 16 + r15;
  const float bi = bias[0 * kH + j];
  const float bf = bias[1 * kH + j];
  const float bg = bias[2 * kH + j];
  const float bo = bias[3 * kH + j];
  float wsi = 0.f, wsf = 0.f, wsg = 0.f, wso = 0.f;
  if (wih_scalar) {
    wsi = wih_scalar[0 * kH + j];
    wsf = wih_scalar[1 * kH + j];
    wsg = wih_scalar[2 * kH + j];
    wso = wih_scalar[3 * kH + j];
  }
  const int hi8 = (lane >> 4) * 8;
#pragma unroll
  for (int e = 0; e < 8; ++e) {
    const int m = mt * 16 + hi8 + e;
    float gi = acc0[e] + bi;
    float gf = acc1[e] + bf;
    float gg = acc2[e] + bg;
    float go = acc3[e] + bo;
    if (xscalar) {
      const float xv = xscalar[m];
      gi += xv * wsi; gf += xv * wsf; gg += xv * wsg; go += xv * wso;
    }
    const size_t idx = (size_t)m * kH + j;
    const float c_old = Cst[idx];
    const float c2 = sigmoidf_(gf) * c_old + sigmoidf_(gi) * tanhf(gg);
    const float h2 = sigmoidf_(go) * tanhf(c2);
    Cst[idx]  = c2;
    Hout[idx] = f2bf(h2);
  }
}

// ------------------------------ fc projection ------------------------------
// out[b, hor] = h1[b,:] . fc_W + fc_b ; also feeds xdec for next decoder step.
__launch_bounds__(256)
__global__ void fc_out(const unsigned short* __restrict__ H1,  // bf16 [kB,kH]
                       const float* __restrict__ fcW,          // f32 [kH]
                       const float* __restrict__ fcb,          // f32 [1]
                       float* __restrict__ out,                // f32 [kB,kHor]
                       float* __restrict__ xdec,               // f32 [kB]
                       int hor) {
  const int lane = threadIdx.x & 31;
  const int row  = blockIdx.x * (blockDim.x >> 5) + (threadIdx.x >> 5);
  float s = 0.f;
  for (int k = lane; k < kH; k += 32)
    s += bf2f(H1[(size_t)row * kH + k]) * fcW[k];
#pragma unroll
  for (int off = 16; off > 0; off >>= 1) s += __shfl_down(s, off, 32);
  if (lane == 0) {
    const float v = s + fcb[0];
    out[(size_t)row * kHor + hor] = v;
    xdec[row] = v;
  }
}

// ------------------------------ host launcher ------------------------------

extern "C" void kernel_launch(void* const* d_in, const int* in_sizes, int n_in,
                              void* d_out, int out_size, void* d_ws, size_t ws_size,
                              hipStream_t stream) {
  (void)in_sizes; (void)n_in; (void)out_size; (void)ws_size;

  const float* x        = (const float*)d_in[0];
  // d_in[1] = y (unused by reference output)
  const float* eWih0 = (const float*)d_in[2];
  const float* eWhh0 = (const float*)d_in[3];
  const float* eb0   = (const float*)d_in[4];
  const float* eWih1 = (const float*)d_in[5];
  const float* eWhh1 = (const float*)d_in[6];
  const float* eb1   = (const float*)d_in[7];
  const float* dWih0 = (const float*)d_in[8];   // [2048,1] -> scalar path
  const float* dWhh0 = (const float*)d_in[9];
  const float* db0   = (const float*)d_in[10];
  const float* dWih1 = (const float*)d_in[11];
  const float* dWhh1 = (const float*)d_in[12];
  const float* db1   = (const float*)d_in[13];
  const float* fcW   = (const float*)d_in[14];
  const float* fcb   = (const float*)d_in[15];
  float* out = (float*)d_out;

  // ---- workspace carve-out ----
  size_t off = 0;
  char* base = (char*)d_ws;
  auto carve = [&](size_t bytes) -> void* {
    void* p = base + off;
    off += (bytes + 255) & ~(size_t)255;
    return p;
  };
  unsigned short* xbf   = (unsigned short*)carve((size_t)kB * kT * kIn * 2);
  unsigned short* wih0  = (unsigned short*)carve((size_t)kG4 * kIn * 2);
  unsigned short* whh0  = (unsigned short*)carve((size_t)kG4 * kH * 2);
  unsigned short* wih1  = (unsigned short*)carve((size_t)kG4 * kH * 2);
  unsigned short* whh1  = (unsigned short*)carve((size_t)kG4 * kH * 2);
  unsigned short* dwhh0 = (unsigned short*)carve((size_t)kG4 * kH * 2);
  unsigned short* dwih1 = (unsigned short*)carve((size_t)kG4 * kH * 2);
  unsigned short* dwhh1 = (unsigned short*)carve((size_t)kG4 * kH * 2);
  unsigned short* h0buf[2] = {(unsigned short*)carve((size_t)kB * kH * 2),
                              (unsigned short*)carve((size_t)kB * kH * 2)};
  unsigned short* h1buf[2] = {(unsigned short*)carve((size_t)kB * kH * 2),
                              (unsigned short*)carve((size_t)kB * kH * 2)};
  float* c0   = (float*)carve((size_t)kB * kH * 4);
  float* c1   = (float*)carve((size_t)kB * kH * 4);
  float* xdec = (float*)carve((size_t)kB * 4);

  auto cvt = [&](const float* src, unsigned short* dst, int n) {
    cvt_f32_to_bf16<<<(n + 255) / 256, 256, 0, stream>>>(src, dst, n);
  };
  auto zero = [&](void* p, size_t dwords) {
    zero_u32<<<((int)dwords + 255) / 256, 256, 0, stream>>>((unsigned*)p, (int)dwords);
  };

  // ---- weight / input conversion to bf16 ----
  cvt(x,     xbf,   kB * kT * kIn);
  cvt(eWih0, wih0,  kG4 * kIn);
  cvt(eWhh0, whh0,  kG4 * kH);
  cvt(eWih1, wih1,  kG4 * kH);
  cvt(eWhh1, whh1,  kG4 * kH);
  cvt(dWhh0, dwhh0, kG4 * kH);
  cvt(dWih1, dwih1, kG4 * kH);
  cvt(dWhh1, dwhh1, kG4 * kH);

  // ---- zero initial state (h bf16, c f32, decoder feedback input) ----
  zero(h0buf[0], (size_t)kB * kH / 2);
  zero(h1buf[0], (size_t)kB * kH / 2);
  zero(c0,       (size_t)kB * kH);
  zero(c1,       (size_t)kB * kH);
  zero(xdec,     (size_t)kB);

  const dim3 sgrid(128), sblk(128);
  int p0 = 0, p1 = 0;

  // ---- encoder: 168 timesteps x 2 layers ----
  for (int t = 0; t < kT; ++t) {
    // layer 0: input = x[:, t, :]  (row stride T*IN in bf16 x)
    lstm_step<<<sgrid, sblk, 0, stream>>>(
        xbf + (size_t)t * kIn, kIn, kT * kIn, wih0,
        h0buf[p0], whh0, eb0, nullptr, nullptr, c0, h0buf[p0 ^ 1]);
    p0 ^= 1;
    // layer 1: input = h0 (just produced)
    lstm_step<<<sgrid, sblk, 0, stream>>>(
        h0buf[p0], kH, kH, wih1,
        h1buf[p1], whh1, eb1, nullptr, nullptr, c1, h1buf[p1 ^ 1]);
    p1 ^= 1;
  }

  // ---- decoder: 24 steps, state continues from encoder final (h, c) ----
  for (int s = 0; s < kHor; ++s) {
    // layer 0: rank-1 input (in=1) handled via scalar epilogue
    lstm_step<<<sgrid, sblk, 0, stream>>>(
        nullptr, 0, 0, nullptr,
        h0buf[p0], dwhh0, db0, xdec, dWih0, c0, h0buf[p0 ^ 1]);
    p0 ^= 1;
    lstm_step<<<sgrid, sblk, 0, stream>>>(
        h0buf[p0], kH, kH, dwih1,
        h1buf[p1], dwhh1, db1, nullptr, nullptr, c1, h1buf[p1 ^ 1]);
    p1 ^= 1;
    // projection + feedback
    fc_out<<<kB / 8, 256, 0, stream>>>(h1buf[p1], fcW, fcb, out, xdec, s);
  }
}
